// BIAS_1013612282247
// MI455X (gfx1250) — compile-verified, matrix operand order checked by reading, not run on hardware
//
#include <hip/hip_runtime.h>

typedef __attribute__((ext_vector_type(16))) __bf16 v16bf;
typedef __attribute__((ext_vector_type(8)))  float  v8f;
typedef __attribute__((ext_vector_type(4)))  float  v4f;

#define N_NODES 4096
#define BT      96          // B*T
#define K_ADJ   3
#define HID_    64
#define T_      12
#define B_      8
#define F_OUT_  12
#define SEGS    8           // split-K segments over m (partials stay L2-resident)
#define SEG_M   (N_NODES / SEGS)
#define NCHUNK  (SEG_M / 32)                     // 16 k-chunks per segment
#define PFD     8                                // prefetch distance in chunks
#define SEGSZ   ((size_t)K_ADJ * N_NODES * BT)   // floats per y-partial segment
#define XPACK_ELEMS ((size_t)BT * N_NODES)       // bf16 elements

__device__ inline __bf16 f2bf(float f) { return (__bf16)f; }

__device__ inline v16bf pack16(v4f a, v4f b, v4f c, v4f d) {
  v16bf r;
#pragma unroll
  for (int i = 0; i < 4; ++i) {
    r[i]      = f2bf(a[i]);
    r[4 + i]  = f2bf(b[i]);
    r[8 + i]  = f2bf(c[i]);
    r[12 + i] = f2bf(d[i]);
  }
  return r;
}

// ---------------------------------------------------------------------------
// Kernel 0: pre-pack X (96 x 4096 fp32) into bf16 laid out in WMMA B-fragment
// order: xp[chunk][bt][half][16], chunk = m/32.
// Fragment lane (half h) holds m = m0 + h*8 + {0..7} and m0 + 16 + h*8 + {0..7}.
// ---------------------------------------------------------------------------
__global__ __launch_bounds__(256) void pack_x_bf16(const float* __restrict__ x,
                                                   __bf16* __restrict__ xp) {
  int idx = blockIdx.x * 256 + threadIdx.x;          // 128 chunks * 96 bt * 2 halves
  int h  = idx & 1;
  int bt = (idx >> 1) % BT;
  int c  = idx / (BT * 2);                            // m-chunk index
  const float* px = x + (size_t)bt * N_NODES + c * 32 + h * 8;
  v4f p0 = *(const v4f*)(px);
  v4f p1 = *(const v4f*)(px + 4);
  v4f p2 = *(const v4f*)(px + 16);
  v4f p3 = *(const v4f*)(px + 20);
  *(v16bf*)(xp + (size_t)idx * 16) = pack16(p0, p1, p2, p3);
}

// ---------------------------------------------------------------------------
// Kernel 1: GLU branch + gcn bias.  Dilated 3x3 conv with W=1 input collapses
// to the middle kernel column (kw=1), H offsets {-2,0,+2}.
// out[b,o,n] = a * sigmoid(g) + gcn_b[o]
// ---------------------------------------------------------------------------
__global__ __launch_bounds__(256) void glu_kernel(const float* __restrict__ x,
                                                  const float* __restrict__ w1,
                                                  const float* __restrict__ b1,
                                                  const float* __restrict__ w2,
                                                  const float* __restrict__ b2,
                                                  const float* __restrict__ gcn_b,
                                                  float* __restrict__ out) {
  int idx = blockIdx.x * 256 + threadIdx.x;          // (b,n) pairs: 8*4096
  int b = idx >> 12;
  int n = idx & (N_NODES - 1);
  float xu[6][3], xv[6][3];
#pragma unroll
  for (int c = 0; c < 6; ++c) {
    const float* pu = x + ((size_t)b * T_ + c) * N_NODES;
    const float* pv = x + ((size_t)b * T_ + 6 + c) * N_NODES;
#pragma unroll
    for (int p = 0; p < 3; ++p) {
      int nn = n - 2 + 2 * p;
      bool ok = (nn >= 0) && (nn < N_NODES);
      xu[c][p] = ok ? pu[nn] : 0.f;
      xv[c][p] = ok ? pv[nn] : 0.f;
    }
  }
#pragma unroll
  for (int o = 0; o < F_OUT_; ++o) {
    float a = b1[o], g = b2[o];
#pragma unroll
    for (int c = 0; c < 6; ++c)
#pragma unroll
      for (int p = 0; p < 3; ++p) {
        a += xu[c][p] * w1[((o * 6 + c) * 3 + p) * 3 + 1];
        g += xv[c][p] * w2[((o * 6 + c) * 3 + p) * 3 + 1];
      }
    float sig = 1.f / (1.f + __expf(-g));
    out[((size_t)b * F_OUT_ + o) * N_NODES + n] = a * sig + gcn_b[o];
  }
}

// ---------------------------------------------------------------------------
// Kernel 2: y[k,n,bt] = sum_m adj[k,n,m] * x[bt,m]  via v_wmma_f32_16x16x32_bf16
// Software-pipelined ping-pong: chunk c+1's A (raw fp32) and 6 B-frags are in
// flight while chunk c's wmma chain executes.  Converted A-frags live in
// per-buffer registers (abf[2]) so the cvt has no WAR hazard against in-flight
// wmma.  adj streams once: NT loads + global_prefetch_b8 PFD chunks ahead
// (one 128B line per row per chunk).  Deterministic split-K partial buffers.
// ---------------------------------------------------------------------------
__global__ __launch_bounds__(256) void adj_x_wmma(const float*  __restrict__ adj,
                                                  const __bf16* __restrict__ xpack,
                                                  float* __restrict__ yws) {
  const int tid  = threadIdx.x;
  const int lane = tid & 31;
  const int wave = tid >> 5;
  const int half = lane >> 4;
  const int l16  = lane & 15;

  const int bx  = blockIdx.x;               // 768 = 3k * 8seg * 32 n-blocks
  const int k   = bx >> 8;                  // / 256
  const int r   = bx & 255;
  const int seg = r >> 5;
  const int nb  = r & 31;
  const int n_base = nb * 128 + wave * 16;
  const int n_row  = n_base + l16;          // A-matrix M row for this lane

  v8f acc[6];
#pragma unroll
  for (int t = 0; t < 6; ++t)
#pragma unroll
    for (int i = 0; i < 8; ++i) acc[t][i] = 0.f;

  const float* arow = adj + ((size_t)k * N_NODES + n_row) * N_NODES + half * 8;
  const int c0 = seg * NCHUNK;              // absolute chunk base for this segment

  v4f   araw[2][4];                         // next-A raw fp32
  v16bf abf[2];                             // converted A (per-buffer regs)
  v16bf bfr[2][6];

  auto loadA = [&](int cl, int buf) {
    const float* pa = arow + (size_t)(c0 + cl) * 32;
    araw[buf][0] = __builtin_nontemporal_load((const v4f*)(pa));
    araw[buf][1] = __builtin_nontemporal_load((const v4f*)(pa + 4));
    araw[buf][2] = __builtin_nontemporal_load((const v4f*)(pa + 16));
    araw[buf][3] = __builtin_nontemporal_load((const v4f*)(pa + 20));
  };
  auto loadB = [&](int cl, int buf) {
#pragma unroll
    for (int t = 0; t < 6; ++t) {
      const int bt = t * 16 + l16;
      bfr[buf][t] =
          *(const v16bf*)(xpack + (((size_t)(c0 + cl) * BT + bt) * 2 + half) * 16);
    }
  };
  auto compute = [&](int buf) {
    abf[buf] = pack16(araw[buf][0], araw[buf][1], araw[buf][2], araw[buf][3]);
#pragma unroll
    for (int t = 0; t < 6; ++t)
      acc[t] = __builtin_amdgcn_wmma_f32_16x16x32_bf16(
          false, abf[buf], false, bfr[buf][t], (short)0, acc[t], false, false);
  };
  auto pf = [&](int cl) {
    int pcl = (cl < NCHUNK) ? cl : NCHUNK - 1;   // clamp: stay inside adj
    __builtin_prefetch((const void*)(arow + (size_t)(c0 + pcl) * 32), 0, 1);
  };

  loadA(0, 0);
  loadB(0, 0);
  // Steady state: unconditional next-chunk addressing (no end-guard selects).
#pragma unroll 1
  for (int c = 0; c < NCHUNK - 2; c += 2) {
    loadA(c + 1, 1);
    loadB(c + 1, 1);
    pf(c + PFD);
    compute(0);
    loadA(c + 2, 0);
    loadB(c + 2, 0);
    pf(c + 1 + PFD);
    compute(1);
  }
  // Peeled tail: chunks NCHUNK-2 and NCHUNK-1.
  loadA(NCHUNK - 1, 1);
  loadB(NCHUNK - 1, 1);
  compute(0);
  compute(1);

  // C layout: lane col N = l16 (bt), rows M = g + 8*half (n offset)
  float* ybase = yws + (size_t)seg * SEGSZ + (size_t)k * N_NODES * BT;
#pragma unroll
  for (int t = 0; t < 6; ++t) {
    const int bt = t * 16 + l16;
#pragma unroll
    for (int g = 0; g < 8; ++g) {
      int n = n_base + g + 8 * half;
      ybase[(size_t)n * BT + bt] = acc[t][g];
    }
  }
}

// ---------------------------------------------------------------------------
// Kernel 3: cheb = relu(y . cheb_w); sgc[b,o,n] = sum_{t,j} cheb * gcn_w[o,t,j]
// WMMA: M = o (padded 12->16), N = (b,n) columns, K = (t,j) = 768.
// B fragments (cheb) generated on the fly from the 36 preloaded y values.
// Dual accumulators (even/odd t) halve the wmma C-RAW dependency chain.
// Adds into d_out on top of the GLU result (stream-ordered, no atomics).
// ---------------------------------------------------------------------------
__global__ __launch_bounds__(256) void cheb_gcn_wmma(const float* __restrict__ yws,
                                                     const float* __restrict__ cheb_w,
                                                     const float* __restrict__ gcn_w,
                                                     float* __restrict__ out) {
  const int tid  = threadIdx.x;
  const int lane = tid & 31;
  const int wave = tid >> 5;
  const int half = lane >> 4;
  const int l16  = lane & 15;

  const int tile = blockIdx.x * 8 + wave;   // 2048 column tiles of 16
  const int col  = tile * 16 + l16;         // col = b*4096 + n
  const int b    = col >> 12;
  const int n    = col & (N_NODES - 1);

  // Preload y[k][t] for this (b,n), summing the SEGS split-K partials.
  float yv[K_ADJ][T_];
#pragma unroll
  for (int k = 0; k < K_ADJ; ++k) {
#pragma unroll
    for (int t = 0; t < T_; ++t) yv[k][t] = 0.f;
#pragma unroll
    for (int s = 0; s < SEGS; ++s) {
      const float* py = yws + (size_t)s * SEGSZ + ((size_t)k * N_NODES + n) * BT + b * T_;
      v4f q0 = *(const v4f*)(py);
      v4f q1 = *(const v4f*)(py + 4);
      v4f q2 = *(const v4f*)(py + 8);
#pragma unroll
      for (int i = 0; i < 4; ++i) {
        yv[k][i]     += q0[i];
        yv[k][4 + i] += q1[i];
        yv[k][8 + i] += q2[i];
      }
    }
  }

  v8f acc0, acc1;
#pragma unroll
  for (int i = 0; i < 8; ++i) { acc0[i] = 0.f; acc1[i] = 0.f; }

  const int o  = l16;                        // A-matrix row (o >= 12 zero-padded)
  const int oc = (o < F_OUT_) ? o : 0;

#pragma unroll
  for (int t = 0; t < T_; ++t) {
#pragma unroll
    for (int jh = 0; jh < 2; ++jh) {
      const int j0 = jh * 32;
      // A fragment: gcn_w[o][t][j]
      const float* pw = gcn_w + ((size_t)oc * T_ + t) * HID_ + j0 + half * 8;
      v4f w0 = *(const v4f*)(pw);
      v4f w1v = *(const v4f*)(pw + 4);
      v4f w2v = *(const v4f*)(pw + 16);
      v4f w3v = *(const v4f*)(pw + 20);
      v16bf af = pack16(w0, w1v, w2v, w3v);
      if (o >= F_OUT_) {
#pragma unroll
        for (int i = 0; i < 16; ++i) af[i] = f2bf(0.f);
      }
      // B fragment: cheb values computed on the fly
      const float y0 = yv[0][t], y1 = yv[1][t], y2 = yv[2][t];
      v16bf bf;
#pragma unroll
      for (int q = 0; q < 8; ++q) {
        int j = j0 + half * 8 + q;
        float c0 = y0 * cheb_w[j * 3 + 0] + y1 * cheb_w[j * 3 + 1] + y2 * cheb_w[j * 3 + 2];
        bf[q] = f2bf(fmaxf(c0, 0.f));
        int j2 = j + 16;
        float c1 = y0 * cheb_w[j2 * 3 + 0] + y1 * cheb_w[j2 * 3 + 1] + y2 * cheb_w[j2 * 3 + 2];
        bf[8 + q] = f2bf(fmaxf(c1, 0.f));
      }
      if (t & 1)
        acc1 = __builtin_amdgcn_wmma_f32_16x16x32_bf16(
            false, af, false, bf, (short)0, acc1, false, false);
      else
        acc0 = __builtin_amdgcn_wmma_f32_16x16x32_bf16(
            false, af, false, bf, (short)0, acc0, false, false);
    }
  }

  // C layout: lane col = l16 -> (b,n); rows o = g + 8*half. o<12 valid.
#pragma unroll
  for (int g = 0; g < 8; ++g) {
    int oo = g + 8 * half;
    if (oo < F_OUT_) {
      size_t idx = ((size_t)b * F_OUT_ + oo) * N_NODES + n;
      out[idx] += acc0[g] + acc1[g];
    }
  }
}

extern "C" void kernel_launch(void* const* d_in, const int* in_sizes, int n_in,
                              void* d_out, int out_size, void* d_ws, size_t ws_size,
                              hipStream_t stream) {
  const float* x      = (const float*)d_in[0];
  const float* adj    = (const float*)d_in[1];
  const float* cheb_w = (const float*)d_in[2];
  const float* gcn_w  = (const float*)d_in[3];
  const float* gcn_b  = (const float*)d_in[4];
  const float* glu1_w = (const float*)d_in[5];
  const float* glu1_b = (const float*)d_in[6];
  const float* glu2_w = (const float*)d_in[7];
  const float* glu2_b = (const float*)d_in[8];
  float* out = (float*)d_out;

  __bf16* xpack = (__bf16*)d_ws;                                   // 768 KB
  float*  yws   = (float*)((char*)d_ws + XPACK_ELEMS * 2);         // 8 x 4.5 MB

  // 0) pack X into bf16 WMMA fragment layout (128 chunks * 96 bt * 2 halves)
  pack_x_bf16<<<(128 * BT * 2) / 256, 256, 0, stream>>>(x, xpack);
  // 1) GLU branch + gcn bias into d_out
  glu_kernel<<<(B_ * N_NODES) / 256, 256, 0, stream>>>(x, glu1_w, glu1_b,
                                                       glu2_w, glu2_b, gcn_b, out);
  // 2) big matmul: 3k * 8seg * 32 n-blocks of 128 rows (8 waves/WG)
  adj_x_wmma<<<K_ADJ * SEGS * 32, 256, 0, stream>>>(adj, xpack, yws);
  // 3) cheb + sgc, added into d_out
  cheb_gcn_wmma<<<2048 / 8, 256, 0, stream>>>(yws, cheb_w, gcn_w, out);
}
